// MPNetForPretraining_78786880077921
// MI455X (gfx1250) — matrix operand order repaired and checked by reading, not compiled, not run on hardware
//
#include <hip/hip_runtime.h>

typedef _Float16 h16;
typedef __attribute__((ext_vector_type(8)))  _Float16 v8h;
typedef __attribute__((ext_vector_type(16))) _Float16 v16h;
typedef __attribute__((ext_vector_type(8)))  float    v8f;

#define E_DIM 768
#define HEADS 12
#define DH 64
#define SKEY 512
#define BATCH 8

#define WMMA_F16(a, b, c) \
  __builtin_amdgcn_wmma_f32_16x16x32_f16(false, (a), false, (b), (short)0, (c), false, false)

__device__ __forceinline__ v16h mk16(v8h lo, v8h hi) {
  v16h r;
#pragma unroll
  for (int j = 0; j < 8; ++j) { r[j] = lo[j]; r[j + 8] = hi[j]; }
  return r;
}

// Row-wise reductions over the 16-lane half that owns each C/D-layout row.
// xor masks {8,4,2,1} never cross the lane-16 boundary (wave32).
__device__ __forceinline__ float rmax16(float v) {
#pragma unroll
  for (int m = 8; m >= 1; m >>= 1) v = fmaxf(v, __shfl_xor(v, m, 32));
  return v;
}
__device__ __forceinline__ float rsum16(float v) {
#pragma unroll
  for (int m = 8; m >= 1; m >>= 1) v += __shfl_xor(v, m, 32);
  return v;
}

// ---------------- conversion kernels ----------------
__global__ void k_cvt(const float* __restrict__ x, h16* __restrict__ y, long n) {
  long i = (long)blockIdx.x * blockDim.x + threadIdx.x;
  if (i < n) y[i] = (h16)x[i];
}

// y[n*768 + k] = (h16)x[k*768 + n]  (weight transpose so B-fragments are contiguous)
__global__ void k_cvt_t(const float* __restrict__ x, h16* __restrict__ y) {
  int i = blockIdx.x * blockDim.x + threadIdx.x;
  if (i >= E_DIM * E_DIM) return;
  int n = i / E_DIM, k = i % E_DIM;
  y[i] = (h16)x[(long)k * E_DIM + n];
}

// ---------------- GEMM: out(MxE) = X(MxE_f16) @ W(ExE) + bias ----------------
// Wt is N-major f16 (row n holds K contiguous). One wave = 32x64 output block:
// 2 M-subtiles x 4 N-subtiles = 8 WMMAs per 32-wide K-step (A reused 4x, B 2x).
// mode 0: f16 -> (B,H,R,D) head layout      (K and Q projections)
// mode 1: f16 -> (B,H,D,S) transposed heads (V projection)
// mode 2: f32 -> row-major d_out            (output projection)
__global__ __launch_bounds__(256) void k_gemm(
    const h16* __restrict__ X, const h16* __restrict__ Wt,
    const float* __restrict__ bias, void* __restrict__ out,
    int M, int mode, float scale, int R, int Sv)
{
  const int tilesM = M >> 5;          // 32-row blocks
  const int tilesN = E_DIM >> 6;      // 12 x 64-col blocks
  int wave = threadIdx.x >> 5;
  int lane = threadIdx.x & 31;
  long tile = (long)blockIdx.x * 8 + wave;
  if (tile >= (long)tilesM * tilesN) return;
  int nt = (int)(tile / tilesM);
  int mt = (int)(tile % tilesM);
  int l15 = lane & 15;
  int koff_a = (lane < 16) ? 0 : 16;   // A frag: contiguous K (k = 16h + s)
  int koff_b = (lane < 16) ? 0 : 16;   // B frag: contiguous K (k = 16h + s), same as A

  const h16* xrow0 = X + (long)(mt * 32 + l15) * E_DIM;
  const h16* xrow1 = xrow0 + 16 * E_DIM;
  const h16* wrow  = Wt + (long)(nt * 64 + l15) * E_DIM;

  v8f acc[2][4];
#pragma unroll
  for (int mi = 0; mi < 2; ++mi)
#pragma unroll
    for (int j = 0; j < 4; ++j) acc[mi][j] = (v8f){};

  for (int k0 = 0; k0 < E_DIM; k0 += 32) {
    v16h a0 = mk16(*(const v8h*)(xrow0 + k0 + koff_a),
                   *(const v8h*)(xrow0 + k0 + koff_a + 8));
    v16h a1 = mk16(*(const v8h*)(xrow1 + k0 + koff_a),
                   *(const v8h*)(xrow1 + k0 + koff_a + 8));
#pragma unroll
    for (int j = 0; j < 4; ++j) {
      v16h b = *(const v16h*)(wrow + (long)j * 16 * E_DIM + k0 + koff_b);
      acc[0][j] = WMMA_F16(a0, b, acc[0][j]);
      acc[1][j] = WMMA_F16(a1, b, acc[1][j]);
    }
  }

  int hi8 = (lane < 16) ? 0 : 8;
#pragma unroll
  for (int mi = 0; mi < 2; ++mi) {
#pragma unroll
    for (int j = 0; j < 4; ++j) {
      int col = nt * 64 + j * 16 + l15;
      float bc = bias[col];
      int hh = col >> 6, dd = col & 63;
#pragma unroll
      for (int r = 0; r < 8; ++r) {
        int m = mt * 32 + mi * 16 + r + hi8;   // flat row = seq*B + batch
        float v = (acc[mi][j][r] + bc) * scale;
        if (mode == 2) {
          ((float*)out)[(long)m * E_DIM + col] = v;
        } else {
          int rr = m >> 3;                     // seq index (B==8)
          int bb = m & 7;                      // batch index
          h16 hv = (h16)v;
          if (mode == 0)
            ((h16*)out)[(((long)bb * HEADS + hh) * R + rr) * DH + dd] = hv;
          else
            ((h16*)out)[(((long)bb * HEADS + hh) * DH + dd) * (long)Sv + rr] = hv;
        }
      }
    }
  }
}

// ---------------- flash attention: one wave per (b, h, 16-query tile) ----------
// 4 waves per block, each with a private 1KB LDS slice for the P-transpose.
__global__ __launch_bounds__(128) void k_attn(
    const h16* __restrict__ Qh,   // (B,H,R,D)  f16, pre-scaled
    const h16* __restrict__ Kh,   // (B,H,S,D)  f16
    const h16* __restrict__ Vt,   // (B,H,D,S)  f16
    const float* __restrict__ bias,           // (H,R,S) f32
    const unsigned char* __restrict__ mask2d, // (R,S) bool
    const unsigned char* __restrict__ kp,     // (B,S) bool
    h16* __restrict__ outp,       // (R*B, E) f16 row-major
    int R, int tilesT)
{
  __shared__ h16 sp_all[4][16 * 32];
  int wave = threadIdx.x >> 5;
  h16* sp = sp_all[wave];
  long wid = (long)blockIdx.x * 4 + wave;   // grids are exact multiples of 4 waves
  int tt = (int)(wid % tilesT);
  int hh = (int)((wid / tilesT) % HEADS);
  int bb = (int)(wid / ((long)tilesT * HEADS));
  int lane = threadIdx.x & 31;
  int l15 = lane & 15;
  int koff_a = (lane < 16) ? 0 : 16;
  int koff_b = (lane < 16) ? 0 : 16;
  int hi8 = (lane < 16) ? 0 : 8;

  // Q A-fragments for K-dim = D (two 32-wide steps)
  const h16* qrow = Qh + (((long)bb * HEADS + hh) * R + tt * 16 + l15) * DH;
  v16h aq0 = mk16(*(const v8h*)(qrow + koff_a),      *(const v8h*)(qrow + koff_a + 8));
  v16h aq1 = mk16(*(const v8h*)(qrow + koff_a + 32), *(const v8h*)(qrow + koff_a + 40));

  const h16* kbase = Kh + ((long)bb * HEADS + hh) * SKEY * DH;
  const h16* vcol  = Vt + (((long)bb * HEADS + hh) * DH + l15) * SKEY;
  const float* brow = bias + ((long)hh * R + tt * 16) * SKEY;
  const unsigned char* mrow = mask2d + (long)(tt * 16) * SKEY;
  const unsigned char* kprow = kp + bb * SKEY;

  float mrun[8], lrun[8];
#pragma unroll
  for (int r = 0; r < 8; ++r) { mrun[r] = -1e30f; lrun[r] = 0.f; }
  v8f o0 = {}, o1 = {}, o2 = {}, o3 = {};

  for (int s0 = 0; s0 < SKEY; s0 += 32) {
    // scores (16 x 32) = Q(16x64) . K^T: 4 WMMAs
    const h16* k0p = kbase + (long)(s0 + l15) * DH;
    const h16* k1p = k0p + 16 * DH;
    if (s0 + 32 < SKEY) {                // prime next chunk (global_prefetch_b8)
      __builtin_prefetch(k0p + 32 * DH, 0, 3);
      __builtin_prefetch(vcol + s0 + 32, 0, 3);
    }
    v8f c0 = {}, c1 = {};
    c0 = WMMA_F16(aq0, *(const v16h*)(k0p + koff_b), c0);
    c0 = WMMA_F16(aq1, *(const v16h*)(k0p + 32 + koff_b), c0);
    c1 = WMMA_F16(aq0, *(const v16h*)(k1p + koff_b), c1);
    c1 = WMMA_F16(aq1, *(const v16h*)(k1p + 32 + koff_b), c1);

    int s_a = s0 + l15, s_b = s_a + 16;
    bool kpa = kprow[s_a] != 0, kpb = kprow[s_b] != 0;
    float x0[8], x1[8];
#pragma unroll
    for (int r = 0; r < 8; ++r) {
      int trow = r + hi8;
      const float* bptr = brow + (long)trow * SKEY;
      const unsigned char* mptr = mrow + (long)trow * SKEY;
      float a0 = c0[r] + bptr[s_a];
      float a1 = c1[r] + bptr[s_b];
      if (mptr[s_a] || kpa) a0 = -1e30f;
      if (mptr[s_b] || kpb) a1 = -1e30f;
      x0[r] = a0; x1[r] = a1;
    }
    // online softmax
    float alpha[8];
#pragma unroll
    for (int r = 0; r < 8; ++r) {
      float cm = rmax16(fmaxf(x0[r], x1[r]));
      float nm = fmaxf(mrun[r], cm);
      alpha[r] = __expf(mrun[r] - nm);
      mrun[r] = nm;
      float p0 = __expf(x0[r] - nm);
      float p1 = __expf(x1[r] - nm);
      x0[r] = p0; x1[r] = p1;
      lrun[r] = lrun[r] * alpha[r] + rsum16(p0 + p1);
    }
#pragma unroll
    for (int r = 0; r < 8; ++r) {
      o0[r] *= alpha[r]; o1[r] *= alpha[r]; o2[r] *= alpha[r]; o3[r] *= alpha[r];
    }
    // transpose P (C/D layout -> A-fragment layout) through LDS
#pragma unroll
    for (int r = 0; r < 8; ++r) {
      sp[(r + hi8) * 32 + l15]      = (h16)x0[r];
      sp[(r + hi8) * 32 + 16 + l15] = (h16)x1[r];
    }
    __syncthreads();
    v16h ap;
#pragma unroll
    for (int j = 0; j < 8; ++j) {
      ap[j]     = sp[l15 * 32 + koff_a + j];
      ap[j + 8] = sp[l15 * 32 + koff_a + 8 + j];
    }
    __syncthreads();
    // O(16x64) += P(16x32) . V(32x64): 4 WMMAs, V K-contiguous via (B,H,D,S)
    const h16* vp = vcol + s0 + koff_b;
    o0 = WMMA_F16(ap, *(const v16h*)(vp),             o0);
    o1 = WMMA_F16(ap, *(const v16h*)(vp + 16 * SKEY), o1);
    o2 = WMMA_F16(ap, *(const v16h*)(vp + 32 * SKEY), o2);
    o3 = WMMA_F16(ap, *(const v16h*)(vp + 48 * SKEY), o3);
  }

  // finalize: divide by row sums and scatter to (t*B+b, h*64+d) f16 rows
#pragma unroll
  for (int r = 0; r < 8; ++r) {
    float inv = 1.0f / lrun[r];
    int t = tt * 16 + r + hi8;
    h16* orow = outp + ((long)t * BATCH + bb) * E_DIM + hh * DH;
    orow[l15]      = (h16)(o0[r] * inv);
    orow[16 + l15] = (h16)(o1[r] * inv);
    orow[32 + l15] = (h16)(o2[r] * inv);
    orow[48 + l15] = (h16)(o3[r] * inv);
  }
}

// ---------------- host ----------------
extern "C" void kernel_launch(void* const* d_in, const int* in_sizes, int n_in,
                              void* d_out, int out_size, void* d_ws, size_t ws_size,
                              hipStream_t stream) {
  (void)in_sizes; (void)n_in; (void)out_size; (void)ws_size;
  const float* c_in  = (const float*)d_in[0];
  const float* q_in  = (const float*)d_in[1];
  const float* key   = (const float*)d_in[2];
  const float* value = (const float*)d_in[3];
  const unsigned char* cmask = (const unsigned char*)d_in[4];
  const unsigned char* qmask = (const unsigned char*)d_in[5];
  const float* cpb = (const float*)d_in[6];
  const float* qpb = (const float*)d_in[7];
  const unsigned char* kpm = (const unsigned char*)d_in[8];
  const float* Wq = (const float*)d_in[9];
  const float* Wk = (const float*)d_in[10];
  const float* Wv = (const float*)d_in[11];
  const float* Wo = (const float*)d_in[12];
  const float* bq = (const float*)d_in[13];
  const float* bk = (const float*)d_in[14];
  const float* bv = (const float*)d_in[15];
  const float* bo = (const float*)d_in[16];

  const long MC = 512L * BATCH;  // 4096 content rows
  const long MQ = 160L * BATCH;  // 1280 query rows

  char* p = (char*)d_ws;
  auto alloc = [&](long bytes) -> char* {
    char* r = p; p += (bytes + 255) & ~255L; return r;
  };
  h16* ch    = (h16*)alloc(MC * E_DIM * 2);
  h16* qh    = (h16*)alloc(MQ * E_DIM * 2);
  h16* keyh  = (h16*)alloc(MC * E_DIM * 2);
  h16* valh  = (h16*)alloc(MC * E_DIM * 2);
  h16* WqT   = (h16*)alloc((long)E_DIM * E_DIM * 2);
  h16* WkT   = (h16*)alloc((long)E_DIM * E_DIM * 2);
  h16* WvT   = (h16*)alloc((long)E_DIM * E_DIM * 2);
  h16* WoT   = (h16*)alloc((long)E_DIM * E_DIM * 2);
  h16* Kh    = (h16*)alloc((long)BATCH * HEADS * SKEY * DH * 2);
  h16* Vt    = (h16*)alloc((long)BATCH * HEADS * DH * SKEY * 2);
  h16* Qc    = (h16*)alloc((long)BATCH * HEADS * 512 * DH * 2);
  h16* Qq    = (h16*)alloc((long)BATCH * HEADS * 160 * DH * 2);
  h16* attnC = (h16*)alloc(MC * E_DIM * 2);
  h16* attnQ = (h16*)alloc(MQ * E_DIM * 2);

  // 1) f16 staging
  k_cvt<<<(int)((MC * E_DIM + 255) / 256), 256, 0, stream>>>(c_in,  ch,   MC * E_DIM);
  k_cvt<<<(int)((MQ * E_DIM + 255) / 256), 256, 0, stream>>>(q_in,  qh,   MQ * E_DIM);
  k_cvt<<<(int)((MC * E_DIM + 255) / 256), 256, 0, stream>>>(key,   keyh, MC * E_DIM);
  k_cvt<<<(int)((MC * E_DIM + 255) / 256), 256, 0, stream>>>(value, valh, MC * E_DIM);
  int wtb = (E_DIM * E_DIM + 255) / 256;
  k_cvt_t<<<wtb, 256, 0, stream>>>(Wq, WqT);
  k_cvt_t<<<wtb, 256, 0, stream>>>(Wk, WkT);
  k_cvt_t<<<wtb, 256, 0, stream>>>(Wv, WvT);
  k_cvt_t<<<wtb, 256, 0, stream>>>(Wo, WoT);

  auto gemm = [&](const h16* X, const h16* Wt, const float* bias, void* out,
                  int M, int mode, float scale, int R, int Sv) {
    long tiles = (long)(M / 32) * (E_DIM / 64);
    int blocks = (int)((tiles + 7) / 8);
    k_gemm<<<blocks, 256, 0, stream>>>(X, Wt, bias, out, M, mode, scale, R, Sv);
  };

  const float scaling = 0.125f;  // 1/sqrt(64)
  // 2) projections into head layouts
  gemm(keyh, WkT, bk, Kh, (int)MC, 0, 1.0f,    512, SKEY);
  gemm(valh, WvT, bv, Vt, (int)MC, 1, 1.0f,    512, SKEY);
  gemm(ch,   WqT, bq, Qc, (int)MC, 0, scaling, 512, SKEY);
  gemm(qh,   WqT, bq, Qq, (int)MQ, 0, scaling, 160, SKEY);

  // 3) attention (flash-style, 4 waves/block, one wave per (b,h,t-tile))
  k_attn<<<BATCH * HEADS * 32 / 4, 128, 0, stream>>>(Qc, Kh, Vt, cpb, cmask, kpm, attnC, 512, 32);
  k_attn<<<BATCH * HEADS * 10 / 4, 128, 0, stream>>>(Qq, Kh, Vt, qpb, qmask, kpm, attnQ, 160, 10);

  // 4) output projection -> fp32 d_out (c_out then q_out, concatenated)
  float* outc = (float*)d_out;
  float* outq = outc + 512L * BATCH * E_DIM;
  gemm(attnC, WoT, bo, outc, (int)MC, 2, 1.0f, 512, SKEY);
  gemm(attnQ, WoT, bo, outq, (int)MQ, 2, 1.0f, 160, SKEY);
}
